// MultiheadAttention_89343909691802
// MI455X (gfx1250) — matrix operand order repaired and checked by reading, not compiled, not run on hardware
//
#include <hip/hip_runtime.h>
#include <hip/hip_bf16.h>

// MI455X (gfx1250, wave32) multihead attention forward.
// GEMMs: v_wmma_f32_16x16x32_f16 with the shared B-panel staged in LDS by the
// Tensor Data Mover (tensor_load_to_lds + s_wait_tensorcnt), B-fragments read
// via ds_load_b128. Attention is computed transposed (S^T = K Q^T) so softmax
// reductions are per-lane + one ds_swizzle SWAPX16.

typedef __attribute__((ext_vector_type(16))) _Float16 v16h;
typedef __attribute__((ext_vector_type(8)))  _Float16 v8h;
typedef __attribute__((ext_vector_type(8)))  float    v8f;
typedef unsigned int u32x4 __attribute__((ext_vector_type(4)));
typedef int          i32x4 __attribute__((ext_vector_type(4)));
typedef int          i32x8 __attribute__((ext_vector_type(8)));

static constexpr int BB   = 4;
static constexpr int TT   = 2048;
static constexpr int CC   = 1024;
static constexpr int HH   = 16;
static constexpr int DD   = 64;
static constexpr int M_T  = BB * TT;        // 8192 rows (b*T + t)
static constexpr int N3C  = 3 * CC;         // 3072

__device__ __forceinline__ v8f wmma16(v16h a, v16h b, v8f c) {
  return __builtin_amdgcn_wmma_f32_16x16x32_f16(
      /*neg_a*/false, a, /*neg_b*/false, b, /*c_mod*/(short)0, c,
      /*reuse_a*/false, /*reuse_b*/false);
}

// A-fragment (16x32 f16, MxK): lane holds row m=lane&15; VGPRs hold
// K = hi*16+[0..15] contiguous  (hi = lane>>4). Two b128 loads.
__device__ __forceinline__ v16h load_a16(const _Float16* rowk0, int hi) {
  const v8h lo = *(const v8h*)(rowk0 + hi * 16);
  const v8h hh = *(const v8h*)(rowk0 + hi * 16 + 8);
  return __builtin_shufflevector(lo, hh, 0,1,2,3,4,5,6,7,8,9,10,11,12,13,14,15);
}

// B-fragment (32x16 f16, KxN) from a column-major (transposed) matrix slice:
// lane holds column n=lane&15 and K = hi*16 + [0..15] contiguous.
__device__ __forceinline__ v16h load_b16(const _Float16* p) {
  const v8h lo = *(const v8h*)(p);
  const v8h hh = *(const v8h*)(p + 8);
  return __builtin_shufflevector(lo, hh, 0,1,2,3,4,5,6,7,8,9,10,11,12,13,14,15);
}

// exchange with lane^16 (SWAPX16 ds_swizzle: xor=0x10, and=0x1f -> 0x401f)
__device__ __forceinline__ float xor16(float v) {
  return __int_as_float(__builtin_amdgcn_ds_swizzle(__float_as_int(v), 0x401f));
}

// ---- Tensor Data Mover: 2D tile (rows x rowlen f16) global -> LDS ----------
// D# per CDNA5 ISA ch.8: group0 {count=1, lds_addr, global_addr, type=2},
// group1 {data_size=2B, tensor dims == tile dims (fully in-bounds),
// tensor_dim0_stride = row pitch}. Remaining groups zero (<=2D tensor).
// This toolchain exposes the 6-arg builtin:
//   (uint32x4 g0, int32x8 g1, int32x4, int32x4, int32x8, i32 cpol)
__device__ __forceinline__ void tdm_load_2d(unsigned lds_byte_off,
                                            const void* gsrc,
                                            unsigned rowlen,   // elements per row
                                            unsigned rows,     // number of rows
                                            unsigned pitch) {  // elements between rows
  const unsigned long long ga = (unsigned long long)gsrc;
  u32x4 g0;
  g0[0] = 1u;                                            // count=1 (valid user D#)
  g0[1] = lds_byte_off;                                  // lds_addr
  g0[2] = (unsigned)(ga & 0xffffffffu);                  // global_addr[31:0]
  g0[3] = (unsigned)((ga >> 32) & 0x01ffffffu) | (2u << 30);  // [56:32] | type=2
  i32x8 g1;
  g1[0] = (int)(1u << 16);                               // data_size=1 -> 2 bytes
  g1[1] = (int)((rowlen & 0xffffu) << 16);               // tensor_dim0[15:0]
  g1[2] = (int)((rowlen >> 16) | ((rows & 0xffffu) << 16));   // d0 hi | d1 lo
  g1[3] = (int)((rows >> 16) | ((rowlen & 0xffffu) << 16));   // d1 hi | tile_dim0
  g1[4] = (int)(rows & 0xffffu);                         // tile_dim1 (tile_dim2=0)
  g1[5] = (int)pitch;                                    // tensor_dim0_stride[31:0]
  g1[6] = 0;                                             // stride hi | dim1_stride lo
  g1[7] = 0;
  const i32x4 z4 = {0, 0, 0, 0};
  const i32x8 z8 = {0, 0, 0, 0, 0, 0, 0, 0};
  __builtin_amdgcn_tensor_load_to_lds(g0, g1, z4, z4, z8, 0);
}

// ---------------- conversion kernels ----------------
__global__ void k_cvt(const float* __restrict__ src, _Float16* __restrict__ dst, int n) {
  int i = blockIdx.x * blockDim.x + threadIdx.x;
  if (i < n) dst[i] = (_Float16)src[i];
}

// src [rows, cols] (row-major) -> dst [cols, rows]  (transpose + f16)
__global__ void k_cvt_t(const float* __restrict__ src, _Float16* __restrict__ dst,
                        int rows, int cols) {
  int i = blockIdx.x * blockDim.x + threadIdx.x;
  if (i < rows * cols) {
    int r = i / cols, c = i % cols;
    dst[(long)c * rows + r] = (_Float16)src[i];
  }
}

// ---- shared GEMM mainloop: block = 8 waves, tile M=128 x N=64, K by 32 ----
// B panel (64 cols x 32 k halves = 4KB) TDM'd into LDS once per k-step and
// consumed by all waves via ds_load_b128. A fragments stream from global.
__device__ __forceinline__ void gemm_mainloop(const _Float16* __restrict__ arow,
                                              const _Float16* __restrict__ BT,
                                              int n0, int K,
                                              _Float16* smem,
                                              int wave, int lane, v8f acc[4]) {
  const int col = lane & 15, hi = lane >> 4;
  const unsigned lds_off = (unsigned)(unsigned long long)(const void*)smem;
  for (int k0 = 0; k0 < K; k0 += 32) {
    if (wave == 0) {
      tdm_load_2d(lds_off, BT + (long)n0 * K + k0, 32u, 64u, (unsigned)K);
      __builtin_amdgcn_s_wait_tensorcnt(0);
    }
    __syncthreads();
    __builtin_prefetch(arow + k0 + 128, 0, 1);
    v16h a = load_a16(arow + k0, hi);
#pragma unroll
    for (int j = 0; j < 4; ++j) {
      // LDS row (j*16+col) of the panel, halves [hi*16, hi*16+16)
      v16h b = load_b16(smem + (j * 16 + col) * 32 + hi * 16);
      acc[j] = wmma16(a, b, acc[j]);
    }
    __syncthreads();
  }
}

// ---------------- QKV projection ----------------
// qkv[M_T, 3C] = Xh @ W_qkv + b ; scatter into Q[bh,t,d]*0.125, K[bh,t,d], Vt[bh,d,t]
__global__ void k_gemm_qkv(const _Float16* __restrict__ Xh,
                           const _Float16* __restrict__ WqkvT,
                           const float* __restrict__ b_qkv,
                           _Float16* __restrict__ Qh,
                           _Float16* __restrict__ Kh,
                           _Float16* __restrict__ Vt) {
  __shared__ __align__(16) _Float16 sB[64 * 32];
  const int lane = threadIdx.x & 31;
  const int wave = threadIdx.x >> 5;
  const int NT4 = N3C / 64;                      // 48
  const int nt = blockIdx.x % NT4;
  const int bm = blockIdx.x / NT4;               // 0..63  (M blocks of 128)
  const int col = lane & 15, hi = lane >> 4;

  const int m0 = bm * 128 + wave * 16;
  const _Float16* arow = Xh + (long)(m0 + col) * CC;
  v8f acc[4] = {};
  gemm_mainloop(arow, WqkvT, nt * 64, CC, sB, wave, lane, acc);

#pragma unroll
  for (int j = 0; j < 4; ++j) {
    const int n = nt * 64 + j * 16 + col;        // column this lane holds
    const float bias = b_qkv[n];
    const int which = n >> 10;                   // 0=q 1=k 2=v
    const int nc = n & (CC - 1);
    const int h = nc >> 6, d = nc & 63;
#pragma unroll
    for (int g = 0; g < 8; ++g) {
      const int m = m0 + g + hi * 8;             // global token row = b*T + t
      const float v = acc[j][g] + bias;
      const int bi = m >> 11;                    // / T
      const int t  = m & (TT - 1);
      const long bh = (long)bi * HH + h;
      if (which == 0)      Qh[(bh * TT + t) * DD + d] = (_Float16)(v * 0.125f);
      else if (which == 1) Kh[(bh * TT + t) * DD + d] = (_Float16)v;
      else                 Vt[(bh * DD + d) * TT + t] = (_Float16)v;
    }
  }
}

// ---------------- causal flash attention (transposed) ----------------
// One wave handles 16 queries x all keys for one (b,h).
// S^T = K_tile(16x64) x Q^T  -> lane owns query column, softmax stats in-lane.
// Y^T = V^T x P^T accumulated in 4 16x16 tiles (D=64).
__global__ void k_attn(const _Float16* __restrict__ Qh,
                       const _Float16* __restrict__ Kh,
                       const _Float16* __restrict__ Vt,
                       _Float16* __restrict__ Yh) {
  const int lane = threadIdx.x & 31;
  const int wave = threadIdx.x >> 5;
  const int widx = blockIdx.x * (blockDim.x >> 5) + wave;  // B*H*(T/16) waves
  const int QT = TT / 16;                                  // 128
  const int bh = widx / QT;
  const int qt = widx % QT;
  if (bh >= BB * HH) return;
  const int q0 = qt * 16;
  const int col = lane & 15, hi = lane >> 4;

  // Q^T B-fragments for the two 32-wide k-steps over D (loaded once)
  const _Float16* qrow = Qh + ((long)bh * TT + q0 + col) * DD;
  const v16h bq0 = load_b16(qrow + 0  + hi * 16);
  const v16h bq1 = load_b16(qrow + 32 + hi * 16);

  v8f y[4] = {};
  float mrun = -INFINITY, lrun = 0.0f;
  const int q = q0 + col;
  const int jend = min(TT, ((q0 + 16 + 31) >> 5) << 5);

  for (int j0 = 0; j0 < jend; j0 += 32) {
    // ---- S^T tiles: tile0 keys [j0,j0+16), tile1 keys [j0+16,j0+32)
    const _Float16* kr0 = Kh + ((long)bh * TT + j0 + col) * DD;
    const _Float16* kr1 = kr0 + 16 * DD;
    v8f s0 = {}, s1 = {};
    s0 = wmma16(load_a16(kr0 + 0,  hi), bq0, s0);
    s0 = wmma16(load_a16(kr0 + 32, hi), bq1, s0);
    s1 = wmma16(load_a16(kr1 + 0,  hi), bq0, s1);
    s1 = wmma16(load_a16(kr1 + 32, hi), bq1, s1);

    // ---- causal mask (lane holds key rows j0+hi*8+g / +16)
    if (j0 + 31 > q0) {
      const int jb = j0 + hi * 8;
#pragma unroll
      for (int g = 0; g < 8; ++g) {
        if (jb + g > q)      s0[g] = -INFINITY;
        if (jb + 16 + g > q) s1[g] = -INFINITY;
      }
    }

    // ---- online softmax (per-query = per-lane + lane^16 merge)
    float tmax = -INFINITY;
#pragma unroll
    for (int g = 0; g < 8; ++g) tmax = fmaxf(tmax, fmaxf(s0[g], s1[g]));
    tmax = fmaxf(tmax, xor16(tmax));
    const float mnew  = fmaxf(mrun, tmax);
    const float scale = __expf(mrun - mnew);
#pragma unroll
    for (int dt = 0; dt < 4; ++dt)
#pragma unroll
      for (int g = 0; g < 8; ++g) y[dt][g] *= scale;

    float p0[8], p1[8], rsum = 0.0f;
#pragma unroll
    for (int g = 0; g < 8; ++g) {
      p0[g] = __expf(s0[g] - mnew);
      p1[g] = __expf(s1[g] - mnew);
      rsum += p0[g] + p1[g];
    }
    rsum += xor16(rsum);
    lrun = lrun * scale + rsum;
    mrun = mnew;

    // ---- reshape P^T (C-layout) into a WMMA B-fragment via lane^16 swaps
    v16h bp;
#pragma unroll
    for (int g = 0; g < 8; ++g) {
      const float o0 = xor16(p0[g]);
      const float o1 = xor16(p1[g]);
      bp[g]     = (_Float16)(hi ? o1 : p0[g]);
      bp[8 + g] = (_Float16)(hi ? p1[g] : o0);
    }

    // ---- Y^T += V^T x P^T   (A rows are d, contiguous in Vt)
#pragma unroll
    for (int dt = 0; dt < 4; ++dt) {
      const _Float16* vptr = Vt + ((long)bh * DD + dt * 16 + col) * TT + j0;
      y[dt] = wmma16(load_a16(vptr, hi), bp, y[dt]);
    }
  }

  // ---- epilogue: divide by l, scatter to Yh[b*T+t, h*64+d]
  const float inv = 1.0f / lrun;
  const int bi = bh >> 4, h = bh & 15;
  _Float16* yrow = Yh + ((long)(bi * TT + q0 + col)) * CC + h * DD;
#pragma unroll
  for (int dt = 0; dt < 4; ++dt)
#pragma unroll
    for (int g = 0; g < 8; ++g)
      yrow[dt * 16 + g + 8 * hi] = (_Float16)(y[dt][g] * inv);
}

// ---------------- output projection ----------------
__global__ void k_gemm_out(const _Float16* __restrict__ Yh,
                           const _Float16* __restrict__ WoutT,
                           const float* __restrict__ b_out,
                           float* __restrict__ out) {
  __shared__ __align__(16) _Float16 sB[64 * 32];
  const int lane = threadIdx.x & 31;
  const int wave = threadIdx.x >> 5;
  const int NT4 = CC / 64;                       // 16
  const int nt = blockIdx.x % NT4;
  const int bm = blockIdx.x / NT4;               // 0..63
  const int col = lane & 15, hi = lane >> 4;

  const int m0 = bm * 128 + wave * 16;
  const _Float16* arow = Yh + (long)(m0 + col) * CC;
  v8f acc[4] = {};
  gemm_mainloop(arow, WoutT, nt * 64, CC, sB, wave, lane, acc);

#pragma unroll
  for (int j = 0; j < 4; ++j) {
    const int n = nt * 64 + j * 16 + col;
    const float bias = b_out[n];
#pragma unroll
    for (int g = 0; g < 8; ++g)
      out[(long)(m0 + g + 8 * hi) * CC + n] = acc[j][g] + bias;
  }
}

extern "C" void kernel_launch(void* const* d_in, const int* in_sizes, int n_in,
                              void* d_out, int out_size, void* d_ws, size_t ws_size,
                              hipStream_t stream) {
  const float* x     = (const float*)d_in[0];
  const float* W_qkv = (const float*)d_in[1];
  const float* b_qkv = (const float*)d_in[2];
  const float* W_out = (const float*)d_in[3];
  const float* b_out = (const float*)d_in[4];
  float* out = (float*)d_out;

  // f16 workspace layout (~92 MB)
  _Float16* ws    = (_Float16*)d_ws;
  const long NX   = (long)M_T * CC;        // 8,388,608
  _Float16* Xh    = ws;
  _Float16* WqkvT = Xh + NX;
  _Float16* WoutT = WqkvT + (long)N3C * CC;
  _Float16* Qh    = WoutT + (long)CC * CC;
  _Float16* Kh    = Qh + NX;
  _Float16* Vt    = Kh + NX;
  _Float16* Yh    = Vt + NX;

  k_cvt<<<(int)((NX + 255) / 256), 256, 0, stream>>>(x, Xh, (int)NX);
  k_cvt_t<<<(N3C * CC + 255) / 256, 256, 0, stream>>>(W_qkv, WqkvT, CC, N3C);
  k_cvt_t<<<(CC * CC + 255) / 256, 256, 0, stream>>>(W_out, WoutT, CC, CC);

  // block = 8 waves = tile 128(M) x 64(N); grids divide exactly
  k_gemm_qkv<<<(M_T / 128) * (N3C / 64), 256, 0, stream>>>(Xh, WqkvT, b_qkv, Qh, Kh, Vt);
  // B*H*(T/16) = 8192 waves, 4 waves/block
  k_attn<<<BB * HH * (TT / 16) / 4, 128, 0, stream>>>(Qh, Kh, Vt, Yh);
  k_gemm_out<<<(M_T / 128) * (CC / 64), 256, 0, stream>>>(Yh, WoutT, b_out, out);
}
